// LSTMDecoder_24266565222885
// MI455X (gfx1250) — matrix operand order repaired and checked by reading, not compiled, not run on hardware
//
#include <hip/hip_runtime.h>

typedef unsigned short u16;
typedef __bf16 bf16_t;
typedef bf16_t v16bf __attribute__((ext_vector_type(16)));
typedef float  v8f   __attribute__((ext_vector_type(8)));

#define NB    2048      // batch
#define NH    128       // hidden
#define NT    100       // timesteps (max_len in reference)
#define NIN   518       // x dim
#define NK    576       // x K-dim padded to multiple of 64
#define NPO   544       // gate output N padded to multiple of 16
#define NOUTD 6         // fed-back output dim

union Frag { v16bf v; uint4 q[2]; };

__device__ __forceinline__ u16 f2bf(float f) {
  union { float f; unsigned u; } v; v.f = f;
  unsigned u = v.u;
  unsigned r = u + 0x7fffu + ((u >> 16) & 1u);   // round-to-nearest-even
  return (u16)(r >> 16);
}
__device__ __forceinline__ float sigf(float x) { return 1.0f / (1.0f + __expf(-x)); }

// ---------------------------------------------------------------------------
// WMMA core: one wave computes Y[16 x 16*JT] += X[16 x K] * W[N x K]^T.
// Branch-free, 2x-unrolled K loop with ping-pong fragments so global_load_b128
// for the next K-slice overlaps the current v_wmma_f32_16x16x32_bf16 group.
// Requires K % 64 == 0 (all K here: 576, 512, 256, 128).
// A frag: lane r=lane&15 holds row M=r; elems 0..15 -> K = k0+16*hf .. +15.
// B frag: lane r holds col N=r; elems 0..15 -> K = k0+16*hf .. +15.
// ---------------------------------------------------------------------------
template <int JT>
__device__ __forceinline__ void gemm_acc(const u16* __restrict__ X, int ldx,
                                         const u16* __restrict__ W, int ldw,
                                         int K, int row0, int n0base,
                                         int lane, v8f* acc)
{
  const int r = lane & 15, hf = lane >> 4;
  const u16* xp  = X + (size_t)(row0 + r) * ldx + 16 * hf;
  const u16* wp0 = W + (size_t)(n0base + r) * ldw + 16 * hf;

  Frag a0, a1, b0[JT], b1[JT];
  auto ldA = [&](Frag& a, int k) {
    a.q[0] = *reinterpret_cast<const uint4*>(xp + k);
    a.q[1] = *reinterpret_cast<const uint4*>(xp + k + 8);
  };
  auto ldB = [&](Frag* b, int k) {
#pragma unroll
    for (int j = 0; j < JT; ++j) {
      const u16* p = wp0 + (size_t)j * 16 * ldw + k;
      b[j].q[0] = *reinterpret_cast<const uint4*>(p);
      b[j].q[1] = *reinterpret_cast<const uint4*>(p + 8);
    }
  };

  ldA(a0, 0); ldB(b0, 0);
  for (int k0 = 0; k0 < K; k0 += 64) {
    ldA(a1, k0 + 32); ldB(b1, k0 + 32);
#pragma unroll
    for (int j = 0; j < JT; ++j)
      acc[j] = __builtin_amdgcn_wmma_f32_16x16x32_bf16(
          false, a0.v, false, b0[j].v, (short)0, acc[j], false, false);
    if (k0 + 64 < K) { ldA(a0, k0 + 64); ldB(b0, k0 + 64); }
#pragma unroll
    for (int j = 0; j < JT; ++j)
      acc[j] = __builtin_amdgcn_wmma_f32_16x16x32_bf16(
          false, a1.v, false, b1[j].v, (short)0, acc[j], false, false);
  }
}

// ------------------------- one-time setup kernels --------------------------

__global__ void k_conv_pad(const float* __restrict__ s, u16* __restrict__ d,
                           int rows, int cols, int dcols, int total)
{
  int i = blockIdx.x * 256 + threadIdx.x;
  if (i >= total) return;
  int r = i / dcols, c = i - r * dcols;
  float v = (r < rows && c < cols) ? s[(size_t)r * cols + c] : 0.0f;
  d[i] = f2bf(v);
}

__global__ void k_copy_f32(const float* __restrict__ s, float* __restrict__ d, int n)
{
  int i = blockIdx.x * 256 + threadIdx.x;
  if (i < n) d[i] = s[i];
}

// src [2][B][H] fp32 -> bf16 concat layout dst[m*ldh + colOff + d*128 + h]
__global__ void k_scatter_h(const float* __restrict__ src, u16* __restrict__ dst,
                            int ldh, int colOff)
{
  int i = blockIdx.x * 256 + threadIdx.x;
  if (i >= 2 * NB * NH) return;
  int d = i >> 18, rem = i & 262143, m = rem >> 7, hh = rem & 127;
  dst[(size_t)m * ldh + colOff + d * 128 + hh] = f2bf(src[i]);
}

// x0 = concat(zeros[B,6], h_att_lstm, h_att_gru); fp32 + zero-padded bf16
__global__ void k_init_x(const float* __restrict__ hal, const float* __restrict__ hag,
                         float* __restrict__ xf, u16* __restrict__ xb)
{
  int i = blockIdx.x * 256 + threadIdx.x;
  if (i >= NB * NK) return;
  int m = i / NK, c = i - m * NK;
  float v = 0.0f;
  if (c >= 6 && c < 262)        v = hal[m * 256 + (c - 6)];
  else if (c >= 262 && c < NIN) v = hag[m * 256 + (c - 262)];
  if (c < NIN) xf[(size_t)m * NIN + c] = v;
  xb[i] = f2bf(v);
}

// zero xgb pad cols 544..575 (gate never writes them; they stay 0 all steps)
__global__ void k_zero_pad(u16* __restrict__ xgb)
{
  int i = blockIdx.x * 256 + threadIdx.x;
  if (i >= NB * (NK - NPO)) return;
  int m = i >> 5, c = NPO + (i & 31);
  xgb[(size_t)m * NK + c] = 0;
}

// ------------------------------ per-step kernels ---------------------------

// xg = x * sigmoid(x @ gate_W^T + gate_b); JT=4 covers cols 0..511 (all <518),
// second launch JT=2 with n0fix=512 covers 512..543 (stores masked / zeroed).
template <int JT>
__global__ void k_gate(const u16* __restrict__ xb, const float* __restrict__ xf,
                       const u16* __restrict__ Wb, const float* __restrict__ bias,
                       float* __restrict__ xgf, u16* __restrict__ xgb, int n0fix)
{
  const int lane = threadIdx.x & 31, w = threadIdx.x >> 5;
  const int row0 = blockIdx.y * 128 + w * 16;
  const int n0base = n0fix + blockIdx.x * 64;
  v8f acc[JT] = {};
  gemm_acc<JT>(xb, NK, Wb, NK, NK, row0, n0base, lane, acc);
  const int r = lane & 15, hf = lane >> 4;
#pragma unroll
  for (int j = 0; j < JT; ++j) {
    const int n = n0base + 16 * j + r;
#pragma unroll
    for (int i = 0; i < 8; ++i) {
      const int m = row0 + i + 8 * hf;
      if (n < NIN) {
        float u = acc[j][i] + bias[n];
        float g = xf[(size_t)m * NIN + n] * sigf(u);
        xgf[(size_t)m * NIN + n] = g;
        xgb[(size_t)m * NK + n] = f2bf(g);
      } else {
        xgb[(size_t)m * NK + n] = 0;
      }
    }
  }
}

// next-step x tail: xin[:,6:544) = xg[:,6:544) (fp32 only below 518)
__global__ void k_tail(const float* __restrict__ xgf, const u16* __restrict__ xgb,
                       float* __restrict__ xinf, u16* __restrict__ xinb)
{
  int i = blockIdx.x * 256 + threadIdx.x;
  if (i >= NB * (NPO - 6)) return;
  int m = i / (NPO - 6);
  int c = 6 + (i - m * (NPO - 6));
  xinb[(size_t)m * NK + c] = xgb[(size_t)m * NK + c];
  if (c < NIN) xinf[(size_t)m * NIN + c] = xgf[(size_t)m * NIN + c];
}

// G[d] = X @ Wih[d]^T + h[d] @ Whh[d]^T + bih[d] + bhh[d]   (N=512, z=dir)
__global__ void k_lstm_gemm(const u16* __restrict__ X, int ldx, int Kx,
                            const u16* __restrict__ Wih, long wihStride,
                            const u16* __restrict__ Hb, int ldh,
                            const u16* __restrict__ Whh, long whhStride,
                            const float* __restrict__ bih, const float* __restrict__ bhh,
                            float* __restrict__ G)
{
  const int lane = threadIdx.x & 31, w = threadIdx.x >> 5;
  const int d = blockIdx.z;
  const int row0 = blockIdx.y * 128 + w * 16;
  const int n0base = blockIdx.x * 64;
  v8f acc[4] = {};
  gemm_acc<4>(X, ldx, Wih + (size_t)d * wihStride, Kx, Kx, row0, n0base, lane, acc);
  gemm_acc<4>(Hb + d * 128, ldh, Whh + (size_t)d * whhStride, 128, 128, row0, n0base, lane, acc);
  const int r = lane & 15, hf = lane >> 4;
  const float* bi = bih + d * 512;
  const float* bh = bhh + d * 512;
  float* Gd = G + (size_t)d * NB * 512;
#pragma unroll
  for (int j = 0; j < 4; ++j) {
    const int n = n0base + 16 * j + r;
#pragma unroll
    for (int i = 0; i < 8; ++i) {
      const int m = row0 + i + 8 * hf;
      Gd[(size_t)m * 512 + n] = acc[j][i] + bi[n] + bh[n];
    }
  }
}

// Y[z] = X(+z*xzoff) @ W[z]^T + bias[z]   (GRU gi / gh paths; N mult of 64)
__global__ void k_gemm1(const u16* __restrict__ X, int ldx, int xzoff, int K,
                        const u16* __restrict__ W, long wStride, int ldw,
                        const float* __restrict__ bias, int bStride,
                        float* __restrict__ Y, int Nout)
{
  const int lane = threadIdx.x & 31, w = threadIdx.x >> 5;
  const int z = blockIdx.z;
  const int row0 = blockIdx.y * 128 + w * 16;
  const int n0base = blockIdx.x * 64;
  v8f acc[4] = {};
  gemm_acc<4>(X + z * xzoff, ldx, W + (size_t)z * wStride, ldw, K, row0, n0base, lane, acc);
  const int r = lane & 15, hf = lane >> 4;
  const float* bz = bias + z * bStride;
  float* Yz = Y + (size_t)z * NB * Nout;
#pragma unroll
  for (int j = 0; j < 4; ++j) {
    const int n = n0base + 16 * j + r;
#pragma unroll
    for (int i = 0; i < 8; ++i) {
      const int m = row0 + i + 8 * hf;
      Yz[(size_t)m * Nout + n] = acc[j][i] + bz[n];
    }
  }
}

// LSTM pointwise cell: c' = sig(f)c + sig(i)tanh(g); h = sig(o)tanh(c')
__global__ void k_lstm_elem(const float* __restrict__ G, float* __restrict__ c,
                            u16* __restrict__ hdst, int ldh, int colOff)
{
  int i = blockIdx.x * 256 + threadIdx.x;
  if (i >= 2 * NB * NH) return;
  int d = i >> 18, rem = i & 262143, m = rem >> 7, hh = rem & 127;
  const float* g = G + ((size_t)d * NB + m) * 512;
  float iv = sigf(g[hh]);
  float fv = sigf(g[128 + hh]);
  float gv = tanhf(g[256 + hh]);
  float ov = sigf(g[384 + hh]);
  float* cp = c + ((size_t)d * NB + m) * 128 + hh;
  float cn = fv * (*cp) + iv * gv;
  *cp = cn;
  hdst[(size_t)m * ldh + colOff + d * 128 + hh] = f2bf(ov * tanhf(cn));
}

// GRU pointwise cell
__global__ void k_gru_elem(const float* __restrict__ gi, const float* __restrict__ gh,
                           float* __restrict__ hg, u16* __restrict__ hdst,
                           int ldh, int colOff)
{
  int i = blockIdx.x * 256 + threadIdx.x;
  if (i >= 2 * NB * NH) return;
  int d = i >> 18, rem = i & 262143, m = rem >> 7, hh = rem & 127;
  const float* a = gi + ((size_t)d * NB + m) * 384;
  const float* b = gh + ((size_t)d * NB + m) * 384;
  float r = sigf(a[hh] + b[hh]);
  float z = sigf(a[128 + hh] + b[128 + hh]);
  float n = tanhf(a[256 + hh] + r * b[256 + hh]);
  float* hp = hg + ((size_t)d * NB + m) * 128 + hh;
  float hn = (1.0f - z) * n + z * (*hp);
  *hp = hn;
  hdst[(size_t)m * ldh + colOff + d * 128 + hh] = f2bf(hn);
}

// y = cat @ fc_W^T + fc_b ; write d_out[:,t,:] and next x head
__global__ void k_fc(const u16* __restrict__ cat, const u16* __restrict__ Wfc,
                     const float* __restrict__ fcb, float* __restrict__ out, int t,
                     float* __restrict__ xinf, u16* __restrict__ xinb)
{
  const int lane = threadIdx.x & 31, w = threadIdx.x >> 5;
  const int row0 = blockIdx.x * 128 + w * 16;
  v8f acc[1] = {};
  gemm_acc<1>(cat, 512, Wfc, 512, 512, row0, 0, lane, acc);
  const int r = lane & 15, hf = lane >> 4;
  if (r < NOUTD) {
#pragma unroll
    for (int i = 0; i < 8; ++i) {
      const int m = row0 + i + 8 * hf;
      float y = acc[0][i] + fcb[r];
      out[((size_t)m * NT + t) * NOUTD + r] = y;
      xinf[(size_t)m * NIN + r] = y;
      xinb[(size_t)m * NK + r] = f2bf(y);
    }
  }
}

// ------------------------------- host driver -------------------------------

extern "C" void kernel_launch(void* const* d_in, const int* in_sizes, int n_in,
                              void* d_out, int out_size, void* d_ws, size_t ws_size,
                              hipStream_t stream)
{
  const float* eh     = (const float*)d_in[0];   // encoder_hidden [4,B,H]
  const float* ec     = (const float*)d_in[1];   // encoder_cell   [4,B,H]
  const float* hgru   = (const float*)d_in[2];   // h_gru          [4,B,H]
  const float* hal    = (const float*)d_in[3];   // h_att_lstm     [B,1,256]
  const float* hag    = (const float*)d_in[4];   // h_att_gru      [B,1,256]
  const float* lWih0s = (const float*)d_in[5];   // [2,512,518]
  const float* lWih1s = (const float*)d_in[6];   // [2,512,256]
  const float* lWhhs  = (const float*)d_in[7];   // [2,2,512,128]
  const float* lbih   = (const float*)d_in[8];   // [2,2,512]
  const float* lbhh   = (const float*)d_in[9];
  const float* gWih0s = (const float*)d_in[10];  // [2,384,518]
  const float* gWih1s = (const float*)d_in[11];  // [2,384,256]
  const float* gWhhs  = (const float*)d_in[12];  // [2,2,384,128]
  const float* gbih   = (const float*)d_in[13];  // [2,2,384]
  const float* gbhh   = (const float*)d_in[14];
  const float* gateWs = (const float*)d_in[15];  // [518,518]
  const float* gatebs = (const float*)d_in[16];  // [518]
  const float* fcWs   = (const float*)d_in[17];  // [6,512]
  const float* fcbs   = (const float*)d_in[18];  // [6]
  float* out = (float*)d_out;
  (void)in_sizes; (void)n_in; (void)out_size;

  char* base = (char*)d_ws;
  size_t off = 0;
  auto take16 = [&](size_t e) -> u16*  { u16*  p = (u16*)(base + off);  off = (off + e * 2 + 255) & ~(size_t)255; return p; };
  auto take32 = [&](size_t e) -> float*{ float* p = (float*)(base + off); off = (off + e * 4 + 255) & ~(size_t)255; return p; };

  // bf16 weights (zero-padded; K padded to 576 on the x path)
  u16* gateW = take16((size_t)NPO * NK);           // [544,576]
  u16* lWih0 = take16((size_t)1024 * NK);          // [2*512,576]
  u16* lWih1 = take16((size_t)1024 * 256);
  u16* lWhh  = take16((size_t)2048 * 128);
  u16* gWih0 = take16((size_t)768 * NK);           // [2*384,576]
  u16* gWih1 = take16((size_t)768 * 256);
  u16* gWhh  = take16((size_t)1536 * 128);
  u16* fcW   = take16((size_t)16 * 512);           // [16,512], rows 6..15 zero
  // activations
  u16* xinb  = take16((size_t)NB * NK);
  u16* xgb   = take16((size_t)NB * NK);
  u16* h0L   = take16((size_t)NB * 256);           // lstm layer0 h (bf16, dir-concat)
  u16* h0G   = take16((size_t)NB * 256);           // gru  layer0 h
  u16* catb  = take16((size_t)NB * 512);           // [lstm_l1 h | gru_l1 h]
  float* xinf = take32((size_t)NB * NIN);
  float* xgf  = take32((size_t)NB * NIN);
  float* cL   = take32((size_t)4 * NB * NH);       // lstm cell state fp32
  float* hG   = take32((size_t)4 * NB * NH);       // gru h state fp32
  float* scr  = take32((size_t)3 * NB * 512);      // G(2*B*512) aliased with gi/gh
  if (off > ws_size) return;                       // scratch too small: bail

  float* G  = scr;                                 // [2][B][512]
  float* gi = scr;                                 // [2][B][384] (after G dead)
  float* gh = scr + (size_t)2 * NB * 384;          // [2][B][384]

  auto g1 = [](size_t n) { return (unsigned)((n + 255) / 256); };

  // ---- one-time weight conversion + state init (re-done every call) ----
  k_conv_pad<<<g1((size_t)NPO*NK),256,0,stream>>>(gateWs, gateW, 518,518, NK, NPO*NK);
  k_conv_pad<<<g1((size_t)1024*NK),256,0,stream>>>(lWih0s, lWih0, 1024,518, NK, 1024*NK);
  k_conv_pad<<<g1((size_t)1024*256),256,0,stream>>>(lWih1s, lWih1, 1024,256, 256, 1024*256);
  k_conv_pad<<<g1((size_t)2048*128),256,0,stream>>>(lWhhs,  lWhh,  2048,128, 128, 2048*128);
  k_conv_pad<<<g1((size_t)768*NK),256,0,stream>>>(gWih0s, gWih0, 768,518, NK, 768*NK);
  k_conv_pad<<<g1((size_t)768*256),256,0,stream>>>(gWih1s, gWih1, 768,256, 256, 768*256);
  k_conv_pad<<<g1((size_t)1536*128),256,0,stream>>>(gWhhs,  gWhh,  1536,128, 128, 1536*128);
  k_conv_pad<<<g1((size_t)16*512),256,0,stream>>>(fcWs, fcW, 6,512, 512, 16*512);

  k_copy_f32<<<g1((size_t)4*NB*NH),256,0,stream>>>(ec,   cL, 4*NB*NH);
  k_copy_f32<<<g1((size_t)4*NB*NH),256,0,stream>>>(hgru, hG, 4*NB*NH);
  k_scatter_h<<<g1((size_t)2*NB*NH),256,0,stream>>>(eh,              h0L, 256, 0);
  k_scatter_h<<<g1((size_t)2*NB*NH),256,0,stream>>>(eh + 2*NB*NH,    catb, 512, 0);
  k_scatter_h<<<g1((size_t)2*NB*NH),256,0,stream>>>(hgru,            h0G, 256, 0);
  k_scatter_h<<<g1((size_t)2*NB*NH),256,0,stream>>>(hgru + 2*NB*NH,  catb, 512, 256);
  k_init_x<<<g1((size_t)NB*NK),256,0,stream>>>(hal, hag, xinf, xinb);
  k_zero_pad<<<g1((size_t)NB*(NK-NPO)),256,0,stream>>>(xgb);

  // ---- time loop: stream order realizes the data dependence chain ----
  for (int t = 0; t < NT; ++t) {
    k_gate<4><<<dim3(8,16,1),256,0,stream>>>(xinb, xinf, gateW, gatebs, xgf, xgb, 0);
    k_gate<2><<<dim3(1,16,1),256,0,stream>>>(xinb, xinf, gateW, gatebs, xgf, xgb, 512);
    k_tail<<<g1((size_t)NB*(NPO-6)),256,0,stream>>>(xgf, xgb, xinf, xinb);

    // LSTM layer 0 (input K=576 padded, recurrent K=128)
    k_lstm_gemm<<<dim3(8,16,2),256,0,stream>>>(xgb, NK, NK,
        lWih0, (long)512*NK, h0L, 256, lWhh, (long)512*128,
        lbih, lbhh, G);
    k_lstm_elem<<<g1((size_t)2*NB*NH),256,0,stream>>>(G, cL, h0L, 256, 0);
    // LSTM layer 1 (input = layer0 concat h, K=256)
    k_lstm_gemm<<<dim3(8,16,2),256,0,stream>>>(h0L, 256, 256,
        lWih1, (long)512*256, catb, 512, lWhh + (size_t)2*512*128, (long)512*128,
        lbih + 1024, lbhh + 1024, G);
    k_lstm_elem<<<g1((size_t)2*NB*NH),256,0,stream>>>(G, cL + (size_t)2*NB*NH, catb, 512, 0);

    // GRU layer 0: gi = x@Wih^T+bih ; gh = h@Whh^T+bhh
    k_gemm1<<<dim3(6,16,2),256,0,stream>>>(xgb, NK, 0, NK,
        gWih0, (long)384*NK, NK, gbih, 384, gi, 384);
    k_gemm1<<<dim3(6,16,2),256,0,stream>>>(h0G, 256, 128, 128,
        gWhh, (long)384*128, 128, gbhh, 384, gh, 384);
    k_gru_elem<<<g1((size_t)2*NB*NH),256,0,stream>>>(gi, gh, hG, h0G, 256, 0);
    // GRU layer 1
    k_gemm1<<<dim3(6,16,2),256,0,stream>>>(h0G, 256, 0, 256,
        gWih1, (long)384*256, 256, gbih + 768, 384, gi, 384);
    k_gemm1<<<dim3(6,16,2),256,0,stream>>>(catb + 256, 512, 128, 128,
        gWhh + (size_t)2*384*128, (long)384*128, 128, gbhh + 768, 384, gh, 384);
    k_gru_elem<<<g1((size_t)2*NB*NH),256,0,stream>>>(gi, gh, hG + (size_t)2*NB*NH, catb, 512, 256);

    // FC: y -> d_out[:,t,:] and next x head
    k_fc<<<dim3(16,1,1),256,0,stream>>>(catb, fcW, fcbs, out, t, xinf, xinb);
  }
}
